// BoardEmbeddingWithTopology_78280073937241
// MI455X (gfx1250) — matrix operand order repaired and checked by reading, not compiled, not run on hardware
//
#include <hip/hip_runtime.h>
#include <stdint.h>

// ---------------------------------------------------------------------------
// CDNA5 (gfx1250) chess-transformer block.
// GEMM math: v_wmma_f32_16x16x32_bf16 (wave32 WMMA, f32 accum).
// Direct-copy LDS stages: global_load_async_to_lds_b128 (ASYNCcnt path).
// ---------------------------------------------------------------------------

typedef __attribute__((ext_vector_type(16))) __bf16        v16bf;
typedef __attribute__((ext_vector_type(8)))  float         v8f;
typedef __attribute__((ext_vector_type(8)))  unsigned int  v8u;

static __device__ __forceinline__ unsigned short f2bf(float f) {
  unsigned int u = __builtin_bit_cast(unsigned int, f);
  u += 0x7FFFu + ((u >> 16) & 1u);            // round-to-nearest-even
  return (unsigned short)(u >> 16);
}

// Async global->LDS 16B copy (per lane). LDS address = low 32 bits of flat ptr
// (ISA 10.2: LDS aperture truncates addr[31:0]).
static __device__ __forceinline__ void async_g2l_b128(const void* g, void* l) {
  unsigned lds = (unsigned)(size_t)l;
  asm volatile("global_load_async_to_lds_b128 %0, %1, off"
               :: "v"(lds), "v"(g) : "memory");
}
static __device__ __forceinline__ void wait_async0() {
  asm volatile("s_wait_asynccnt 0" ::: "memory");
}

// A-matrix 16x32 bf16 fragment (ISA 7.12.2): lane(0-15)=row M, khalf=lane>>4,
// element K = khalf*8 + {0..7} and khalf*8 + 16 + {0..7}  -> two 16B LDS loads.
static __device__ __forceinline__ v16bf frag_a(const unsigned short* row, int khalf) {
  const uint4 a0 = *(const uint4*)(row + khalf * 8);
  const uint4 a1 = *(const uint4*)(row + khalf * 8 + 16);
  v8u t;
  t[0] = a0.x; t[1] = a0.y; t[2] = a0.z; t[3] = a0.w;
  t[4] = a1.x; t[5] = a1.y; t[6] = a1.z; t[7] = a1.w;
  return __builtin_bit_cast(v16bf, t);
}

// B-matrix 32x16 bf16 fragment: lane(0-15)=col N, khalf=lane>>4, element
// K = khalf*8 + {0..7} and khalf*8 + 16 + {0..7} (same as A); B staged [n][k] -> two 16B loads.
static __device__ __forceinline__ v16bf frag_b(const unsigned short* row, int khalf) {
  const uint4 a0 = *(const uint4*)(row + khalf * 8);
  const uint4 a1 = *(const uint4*)(row + khalf * 8 + 16);
  v8u t;
  t[0] = a0.x; t[1] = a0.y; t[2] = a0.z; t[3] = a0.w;
  t[4] = a1.x; t[5] = a1.y; t[6] = a1.z; t[7] = a1.w;
  return __builtin_bit_cast(v16bf, t);
}

static __device__ __forceinline__ v8f wmma_bf16(v16bf a, v16bf b, v8f c) {
  return __builtin_amdgcn_wmma_f32_16x16x32_bf16(false, a, false, b, (short)0, c,
                                                 false, false);
}

// ---------------------------------------------------------------------------
// Weight conversion kernels (f32 -> bf16), emb_W zero-padded K 119 -> 128.
// ---------------------------------------------------------------------------
__global__ __launch_bounds__(256) void cvt_bf16_kernel(const float* __restrict__ src,
                                                       unsigned short* __restrict__ dst,
                                                       int n) {
  int i = blockIdx.x * 256 + threadIdx.x;
  if (i < n) dst[i] = f2bf(src[i]);
}

__global__ __launch_bounds__(256) void cvt_embw_kernel(const float* __restrict__ src,
                                                       unsigned short* __restrict__ dst) {
  int i = blockIdx.x * 256 + threadIdx.x;   // over 128*1024
  int r = i >> 10, c = i & 1023;
  dst[i] = (r < 119) ? f2bf(src[r * 1024 + c]) : (unsigned short)0;
}

// ---------------------------------------------------------------------------
// Tiled WMMA GEMM: C[M,1024] = A[M,K] * W[K,1024] (+bias, +epilogue)
// BM=128, BN=128, BK=32. 256 threads = 8 waves; wave w owns M-tiles 2*(w>>1),
// N-tiles (w&1)*4 .. +3  -> wave tile 32x64 = 8 x v8f accumulators
// (8 WMMA per stage vs 6 fragment loads).
// AMODE: 0 = im2col of board (f32, zero-pad K>=119), 1 = f32 row-major,
//        2 = bf16 row-major (direct copy -> async global->LDS).
// EMODE: 0 = f32 out, + bias + positional[(row%64)]   (embedding)
//        1 = bf16 out, (acc+bias)*scale               (q/k/v)
//        2 = f32 out, + bias + residual[row]          (o-proj + residual)
// ---------------------------------------------------------------------------
template <int AMODE, int EMODE>
__global__ __launch_bounds__(256) void gemm_tile(const void* __restrict__ Aptr,
                                                 const unsigned short* __restrict__ Wb,
                                                 const float* __restrict__ bias,
                                                 const float* __restrict__ extra,
                                                 void* __restrict__ outp,
                                                 int K, float scale) {
  __shared__ unsigned short As[128][32];   // [m][k] bf16 (8 KB)
  __shared__ unsigned short Bs[128][32];   // [n][k] bf16 transposed stage (8 KB)

  const int tid  = threadIdx.x;
  const int lane = tid & 31;
  const int wv   = tid >> 5;
  const int row0 = blockIdx.y * 128;
  const int n0   = blockIdx.x * 128;
  const int mt2  = wv >> 1;            // 2 consecutive 16-row tiles
  const int nt0  = (wv & 1) * 4;
  const int l15  = lane & 15;
  const int kh   = lane >> 4;

  const v8f vzero = {0.f, 0.f, 0.f, 0.f, 0.f, 0.f, 0.f, 0.f};
  v8f acc[2][4] = {{vzero, vzero, vzero, vzero}, {vzero, vzero, vzero, vzero}};

  for (int k0 = 0; k0 < K; k0 += 32) {
    // ---- stage A (128x32) ----
    if (AMODE == 2) {
      // direct bf16 copy: async DMA global -> LDS, 2 x 16B per thread
      const unsigned short* A = (const unsigned short*)Aptr;
      int r = tid >> 1, c = (tid & 1) * 16;
      const unsigned short* gp = A + (size_t)(row0 + r) * 1024 + k0 + c;
      async_g2l_b128(gp,     &As[r][c]);
      async_g2l_b128(gp + 8, &As[r][c + 8]);
    } else {
#pragma unroll
      for (int i = 0; i < 16; ++i) {
        int e = tid + i * 256;
        int r = e >> 5, c = e & 31;
        int grow = row0 + r;
        if (AMODE == 0) {
          const float* board = (const float*)Aptr;   // (B, C=119, 64)
          int kk = k0 + c;
          float fv = (kk < 119)
                         ? board[((size_t)(grow >> 6) * 119 + kk) * 64 + (grow & 63)]
                         : 0.f;
          As[r][c] = f2bf(fv);
        } else {
          const float* A = (const float*)Aptr;
          As[r][c] = f2bf(A[(size_t)grow * 1024 + k0 + c]);
        }
      }
    }
    // ---- stage B (32x128) transposed into Bs[n][k] ----
#pragma unroll
    for (int i = 0; i < 16; ++i) {
      int e = tid + i * 256;
      int r = e >> 7, c = e & 127;
      Bs[c][r] = Wb[(size_t)(k0 + r) * 1024 + n0 + c];
    }
    if (AMODE == 2) wait_async0();
    __syncthreads();

    v16bf af0 = frag_a(&As[mt2 * 32 + l15][0], kh);
    v16bf af1 = frag_a(&As[mt2 * 32 + 16 + l15][0], kh);
#pragma unroll
    for (int j = 0; j < 4; ++j) {
      v16bf bfr = frag_b(&Bs[(nt0 + j) * 16 + l15][0], kh);
      acc[0][j] = wmma_bf16(af0, bfr, acc[0][j]);
      acc[1][j] = wmma_bf16(af1, bfr, acc[1][j]);
    }
    __syncthreads();
  }

  // ---- epilogue: C/D layout m = vgpr + (lane>>4)*8, n = lane&15 ----
#pragma unroll
  for (int mi = 0; mi < 2; ++mi) {
#pragma unroll
    for (int j = 0; j < 4; ++j) {
#pragma unroll
      for (int vv = 0; vv < 8; ++vv) {
        int m    = mt2 * 32 + mi * 16 + vv + kh * 8;
        int n    = n0 + (nt0 + j) * 16 + l15;
        int grow = row0 + m;
        float val = acc[mi][j][vv] + bias[n];
        if (EMODE == 0) {
          ((float*)outp)[(size_t)grow * 1024 + n] =
              val + extra[(size_t)(grow & 63) * 1024 + n];
        } else if (EMODE == 1) {
          ((unsigned short*)outp)[(size_t)grow * 1024 + n] = f2bf(val * scale);
        } else {
          ((float*)outp)[(size_t)grow * 1024 + n] =
              val + extra[(size_t)grow * 1024 + n];
        }
      }
    }
  }
}

// ---------------------------------------------------------------------------
// Attention: one workgroup per (batch, head). q pre-scaled by 1/8.
// S = q k^T + rel_bias[h]; softmax rows in f32; O = P V. All GEMMs via WMMA.
// q/k/v stored as [B,64,H,64] bf16 (== row-major [M,1024] with n = h*64+d).
// q/k tile loads are direct copies -> async global->LDS DMA.
// ---------------------------------------------------------------------------
__global__ __launch_bounds__(256) void attention_kernel(
    const unsigned short* __restrict__ qg, const unsigned short* __restrict__ kg,
    const unsigned short* __restrict__ vg, const float* __restrict__ rel_bias,
    unsigned short* __restrict__ og) {
  __shared__ unsigned short qs[64][64];   // [s][d]
  __shared__ unsigned short ks[64][64];   // [s][d]
  __shared__ unsigned short vt[64][64];   // transposed: [d][s]
  __shared__ float          Ss[64][64];   // f32 scores
  __shared__ unsigned short Ps[64][64];   // bf16 probabilities

  const int tid  = threadIdx.x;
  const int lane = tid & 31;
  const int wv   = tid >> 5;
  const int l15  = lane & 15;
  const int kh   = lane >> 4;
  const int b    = blockIdx.x >> 4;
  const int h    = blockIdx.x & 15;

  // ---- cooperative load: thread t -> row s = t>>2, 16 d's ----
  {
    int s = tid >> 2, d0 = (tid & 3) * 16;
    size_t base = (((size_t)(b * 64 + s) * 16 + h) << 6) + d0;
    async_g2l_b128(qg + base,     &qs[s][d0]);
    async_g2l_b128(qg + base + 8, &qs[s][d0 + 8]);
    async_g2l_b128(kg + base,     &ks[s][d0]);
    async_g2l_b128(kg + base + 8, &ks[s][d0 + 8]);
    unsigned short tmp[16];
    *(uint4*)tmp       = *(const uint4*)(vg + base);
    *(uint4*)(tmp + 8) = *(const uint4*)(vg + base + 8);
#pragma unroll
    for (int i = 0; i < 16; ++i) vt[d0 + i][s] = tmp[i];
    wait_async0();
  }
  __syncthreads();

  const v8f vzero = {0.f, 0.f, 0.f, 0.f, 0.f, 0.f, 0.f, 0.f};

  // ---- scores: 16 tiles of 16x16, 2 per wave; K = DH = 64 (2 wmma) ----
#pragma unroll
  for (int t = 0; t < 2; ++t) {
    int tt = wv * 2 + t, ti = tt >> 2, tj = tt & 3;
    v8f acc = vzero;
#pragma unroll
    for (int kc = 0; kc < 2; ++kc) {
      v16bf af = frag_a(&qs[ti * 16 + l15][kc * 32], kh);
      v16bf bfr = frag_b(&ks[tj * 16 + l15][kc * 32], kh);  // B[k=d][n=j]=k[j][d]
      acc = wmma_bf16(af, bfr, acc);
    }
#pragma unroll
    for (int vv = 0; vv < 8; ++vv) {
      int i = ti * 16 + vv + kh * 8;
      int j = tj * 16 + l15;
      Ss[i][j] = acc[vv] + rel_bias[((size_t)h * 64 + i) * 64 + j];
    }
  }
  __syncthreads();

  // ---- softmax: row = tid>>2, each thread 16 cols; combine over 4 lanes ----
  {
    int r = tid >> 2, c0 = (tid & 3) * 16;
    float e[16];
    float m = -3.4e38f;
#pragma unroll
    for (int i = 0; i < 16; ++i) {
      e[i] = Ss[r][c0 + i];
      m    = fmaxf(m, e[i]);
    }
    m = fmaxf(m, __shfl_xor(m, 1));
    m = fmaxf(m, __shfl_xor(m, 2));
    float s = 0.f;
#pragma unroll
    for (int i = 0; i < 16; ++i) {
      e[i] = __expf(e[i] - m);
      s += e[i];
    }
    s += __shfl_xor(s, 1);
    s += __shfl_xor(s, 2);
    float rinv = __builtin_amdgcn_rcpf(s);
#pragma unroll
    for (int i = 0; i < 16; ++i) Ps[r][c0 + i] = f2bf(e[i] * rinv);
  }
  __syncthreads();

  // ---- O = P @ V : K = 64 keys (2 wmma), B[k=j][n=d] = vt[d][j] ----
#pragma unroll
  for (int t = 0; t < 2; ++t) {
    int tt = wv * 2 + t, ti = tt >> 2, tj = tt & 3;
    v8f acc = vzero;
#pragma unroll
    for (int kc = 0; kc < 2; ++kc) {
      v16bf af = frag_a(&Ps[ti * 16 + l15][kc * 32], kh);
      v16bf bfr = frag_b(&vt[tj * 16 + l15][kc * 32], kh);
      acc = wmma_bf16(af, bfr, acc);
    }
#pragma unroll
    for (int vv = 0; vv < 8; ++vv) {
      int i = ti * 16 + vv + kh * 8;
      int d = tj * 16 + l15;
      og[(((size_t)(b * 64 + i) * 16 + h) << 6) + d] = f2bf(acc[vv]);
    }
  }
}

// ---------------------------------------------------------------------------
// In-place LayerNorm over D=1024: one block per row, 256 threads x float4.
// ---------------------------------------------------------------------------
__global__ __launch_bounds__(256) void layernorm_kernel(float* __restrict__ y,
                                                        const float* __restrict__ g,
                                                        const float* __restrict__ bt) {
  __shared__ float redA[8], redB[8], sh[2];
  const int tid  = threadIdx.x;
  const int lane = tid & 31;
  const int wv   = tid >> 5;
  const size_t row = blockIdx.x;

  float4 v = *(const float4*)(y + row * 1024 + tid * 4);
  float s1 = v.x + v.y + v.z + v.w;
  float s2 = v.x * v.x + v.y * v.y + v.z * v.z + v.w * v.w;
#pragma unroll
  for (int off = 16; off > 0; off >>= 1) {
    s1 += __shfl_xor(s1, off);
    s2 += __shfl_xor(s2, off);
  }
  if (lane == 0) { redA[wv] = s1; redB[wv] = s2; }
  __syncthreads();
  if (tid == 0) {
    float a = 0.f, b2 = 0.f;
    for (int i = 0; i < 8; ++i) { a += redA[i]; b2 += redB[i]; }
    float mu  = a * (1.f / 1024.f);
    float var = b2 * (1.f / 1024.f) - mu * mu;
    sh[0] = mu;
    sh[1] = rsqrtf(var + 1e-5f);
  }
  __syncthreads();
  float mu = sh[0], rs = sh[1];
  float4 gg = *(const float4*)(g + tid * 4);
  float4 bb = *(const float4*)(bt + tid * 4);
  float4 o;
  o.x = (v.x - mu) * rs * gg.x + bb.x;
  o.y = (v.y - mu) * rs * gg.y + bb.y;
  o.z = (v.z - mu) * rs * gg.z + bb.z;
  o.w = (v.w - mu) * rs * gg.w + bb.w;
  *(float4*)(y + row * 1024 + tid * 4) = o;
}

// ---------------------------------------------------------------------------
extern "C" void kernel_launch(void* const* d_in, const int* in_sizes, int n_in,
                              void* d_out, int out_size, void* d_ws, size_t ws_size,
                              hipStream_t stream) {
  (void)in_sizes; (void)n_in; (void)out_size; (void)ws_size;

  const float* board      = (const float*)d_in[0];
  const float* emb_W      = (const float*)d_in[1];
  const float* emb_b      = (const float*)d_in[2];
  const float* positional = (const float*)d_in[3];
  const float* Wq = (const float*)d_in[4];
  const float* bq = (const float*)d_in[5];
  const float* Wk = (const float*)d_in[6];
  const float* bk = (const float*)d_in[7];
  const float* Wv = (const float*)d_in[8];
  const float* bv = (const float*)d_in[9];
  const float* Wo = (const float*)d_in[10];
  const float* bo = (const float*)d_in[11];
  const float* rel_bias = (const float*)d_in[12];
  const float* ln_g = (const float*)d_in[13];
  const float* ln_b = (const float*)d_in[14];

  const size_t M = 1024ull * 64, D = 1024;
  unsigned char* ws = (unsigned char*)d_ws;
  size_t off = 0;
  float*          x    = (float*)(ws + off);          off += M * D * 4;   // x (f32)
  unsigned short* qb   = (unsigned short*)(ws + off); off += M * D * 2;   // q bf16 (pre-scaled)
  unsigned short* kb   = (unsigned short*)(ws + off); off += M * D * 2;
  unsigned short* vb   = (unsigned short*)(ws + off); off += M * D * 2;
  unsigned short* ob   = (unsigned short*)(ws + off); off += M * D * 2;   // attn out bf16
  unsigned short* eWb  = (unsigned short*)(ws + off); off += 128 * 1024 * 2;
  unsigned short* Wqb  = (unsigned short*)(ws + off); off += 1024 * 1024 * 2;
  unsigned short* Wkb  = (unsigned short*)(ws + off); off += 1024 * 1024 * 2;
  unsigned short* Wvb  = (unsigned short*)(ws + off); off += 1024 * 1024 * 2;
  unsigned short* Wob  = (unsigned short*)(ws + off); off += 1024 * 1024 * 2;

  // --- weight conversion to bf16 ---
  cvt_embw_kernel<<<(128 * 1024) / 256, 256, 0, stream>>>(emb_W, eWb);
  cvt_bf16_kernel<<<(1024 * 1024) / 256, 256, 0, stream>>>(Wq, Wqb, 1024 * 1024);
  cvt_bf16_kernel<<<(1024 * 1024) / 256, 256, 0, stream>>>(Wk, Wkb, 1024 * 1024);
  cvt_bf16_kernel<<<(1024 * 1024) / 256, 256, 0, stream>>>(Wv, Wvb, 1024 * 1024);
  cvt_bf16_kernel<<<(1024 * 1024) / 256, 256, 0, stream>>>(Wo, Wob, 1024 * 1024);

  dim3 gg(8, 512);  // N/128 x M/128

  // --- embedding: x = transpose(board) @ emb_W + emb_b + positional ---
  gemm_tile<0, 0><<<gg, 256, 0, stream>>>(board, eWb, emb_b, positional, x, 128, 1.0f);

  // --- q/k/v projections (q pre-scaled by 1/sqrt(64) = 0.125, exact in bf16) ---
  gemm_tile<1, 1><<<gg, 256, 0, stream>>>(x, Wqb, bq, nullptr, qb, 1024, 0.125f);
  gemm_tile<1, 1><<<gg, 256, 0, stream>>>(x, Wkb, bk, nullptr, kb, 1024, 1.0f);
  gemm_tile<1, 1><<<gg, 256, 0, stream>>>(x, Wvb, bv, nullptr, vb, 1024, 1.0f);

  // --- attention: one block per (b, h) ---
  attention_kernel<<<1024 * 16, 256, 0, stream>>>(qb, kb, vb, rel_bias, ob);

  // --- output projection + bias + residual into d_out (f32) ---
  gemm_tile<2, 2><<<gg, 256, 0, stream>>>(ob, Wob, bo, x, d_out, 1024, 1.0f);

  // --- in-place LayerNorm over D=1024 ---
  layernorm_kernel<<<(unsigned)M, 256, 0, stream>>>((float*)d_out, ln_g, ln_b);
}
